// TCSAModule_44341242363919
// MI455X (gfx1250) — matrix operand order repaired and checked by reading, not compiled, not run on hardware
//
#include <hip/hip_runtime.h>
#include <hip/hip_bf16.h>

// Fixed problem sizes from the reference
#define NN 32
#define CC 128
#define HH 32
#define WW 32
#define TT 12
#define RR 64
#define HWp (HH*WW)          // 1024
#define CT  (HWp*TT)         // 12288 elements per (n,c) slab
#define JTOT (HWp*TT)        // fused (s,t) column dimension = 12288

typedef __attribute__((ext_vector_type(16))) _Float16 v16h;
typedef __attribute__((ext_vector_type(8)))  float    v8f;

// ---------------------------------------------------------------------------
// Kernel A: pooled[n,t,c] = mean_hw x[n,c,hw,t]
// block = one (n,c); 384 threads (12*32): lane tid has fixed t = tid%12 and
// reads fully contiguous 384-float chunks (stride 384 per iter) -> coalesced.
// ---------------------------------------------------------------------------
__global__ void k_pooled(const float* __restrict__ x, float* __restrict__ pooled) {
    const int nc = blockIdx.x;               // n*128 + c
    const int n  = nc >> 7;
    const int c  = nc & 127;
    const float* px = x + (size_t)nc * CT;
    const int tid = threadIdx.x;             // 0..383
    float acc = 0.f;
#pragma unroll
    for (int i = 0; i < 32; ++i) acc += px[i * 384 + tid];
    __shared__ float lds[384];
    lds[tid] = acc;
    __syncthreads();
    if (tid < 12) {
        float s = 0.f;
#pragma unroll
        for (int g = 0; g < 32; ++g) s += lds[tid + 12 * g];
        pooled[(n * TT + tid) * CC + c] = s * (1.0f / (float)HWp);
    }
}

// ---------------------------------------------------------------------------
// Kernel B: ts_in[n,t,hw] = mean_c x[n,c,hw,t]
// block = (n, chunk of 32 hw); 384 threads own one (hw,t) each; per-c reads
// are 384 consecutive floats -> coalesced. 128 iterations over c.
// ---------------------------------------------------------------------------
__global__ void k_tsin(const float* __restrict__ x, float* __restrict__ ts_in) {
    const int b = blockIdx.x;
    const int n = b >> 5;
    const int chunk = b & 31;
    const int tid = threadIdx.x;             // 0..383
    const float* px = x + (size_t)n * CC * CT + chunk * 384;
    float acc = 0.f;
#pragma unroll 8
    for (int c = 0; c < CC; ++c) acc += px[(size_t)c * CT + tid];
    const int hw = chunk * 32 + tid / 12;
    const int t  = tid % 12;
    ts_in[(n * TT + t) * HWp + hw] = acc * (1.0f / (float)CC);
}

// ---------------------------------------------------------------------------
// Kernel C: TS[n,r,hw] = sigmoid( conv3x3(ts_in)[n,r,hw] + ts_b[r] )
// block = (n, r); weight indices are wave-uniform -> scalar loads; ts_in is
// tiny (1.5 MB) and L2/L0 resident.
// ---------------------------------------------------------------------------
__global__ void k_conv_sig(const float* __restrict__ ts_in,
                           const float* __restrict__ ts_w,
                           const float* __restrict__ ts_b,
                           float* __restrict__ TS) {
    const int b = blockIdx.x;
    const int n = b >> 6;
    const int r = b & 63;
    const float* w  = ts_w + r * (TT * 9);
    const float  bias = ts_b[r];
    const float* in = ts_in + n * TT * HWp;
#pragma unroll
    for (int o = 0; o < 4; ++o) {
        const int hw = threadIdx.x + o * 256;        // 0..1023
        const int h = hw >> 5, wq = hw & 31;
        float acc = bias;
        for (int t = 0; t < TT; ++t) {
#pragma unroll
            for (int ky = 0; ky < 3; ++ky) {
                const int hh = h + ky - 1;
                if (hh < 0 || hh >= HH) continue;
#pragma unroll
                for (int kx = 0; kx < 3; ++kx) {
                    const int ww2 = wq + kx - 1;
                    if (ww2 < 0 || ww2 >= WW) continue;
                    acc += in[t * HWp + (hh << 5) + ww2] * w[t * 9 + ky * 3 + kx];
                }
            }
        }
        TS[((n * RR + r) << 10) + hw] = 1.0f / (1.0f + __expf(-acc));
    }
}

// ---------------------------------------------------------------------------
// Kernel D1: TC[n,t,r] = sigmoid( pooled[n,t,:] . tc_w[r,:] + tc_b[r] )
// ---------------------------------------------------------------------------
__global__ void k_tc(const float* __restrict__ pooled,
                     const float* __restrict__ tc_w,
                     const float* __restrict__ tc_b,
                     float* __restrict__ TC) {
    const int idx = blockIdx.x * 256 + threadIdx.x;  // N*T*R = 24576
    const int nt = idx >> 6;
    const int r  = idx & 63;
    const float* p = pooled + nt * CC;
    const float* w = tc_w + r * CC;
    float acc = tc_b[r];
#pragma unroll 8
    for (int c = 0; c < CC; ++c) acc += p[c] * w[c];
    TC[idx] = 1.0f / (1.0f + __expf(-acc));
}

// ---------------------------------------------------------------------------
// Kernel D2: SC[n,c,r] = sigmoid( pooled[n,:,c] . sc_w[r,:] + sc_b[r] )
// ---------------------------------------------------------------------------
__global__ void k_sc(const float* __restrict__ pooled,
                     const float* __restrict__ sc_w,
                     const float* __restrict__ sc_b,
                     float* __restrict__ SC) {
    const int idx = blockIdx.x * 256 + threadIdx.x;  // N*C*R = 262144
    const int n = idx >> 13;
    const int c = (idx >> 6) & 127;
    const int r = idx & 63;
    float acc = sc_b[r];
#pragma unroll
    for (int t = 0; t < TT; ++t)
        acc += pooled[(n * TT + t) * CC + c] * sc_w[r * TT + t];
    SC[idx] = 1.0f / (1.0f + __expf(-acc));
}

// ---------------------------------------------------------------------------
// Kernel E: the fused CP-reconstruction GEMM + relu(x * M) epilogue.
//   Per n:  M = SC[n] (128x64)  @  B'(64 x 12288),
//           B'[r, j] = TS[n, r, j/12] * TC[n, j%12, r]   (j = s*12 + t)
//   out[(n*128+c)*12288 + j] = relu(M[c,j] * x[same flat index])
// Block = (n, 256-column chunk). 8 waves; wave w owns c-tile w (16 rows).
// A/B fragments staged in LDS pre-permuted to per-lane-contiguous v16h.
// K=64 -> two v_wmma_f32_16x16x32_f16 per 16x16 tile, f32 accumulate.
// ---------------------------------------------------------------------------
__global__ void __launch_bounds__(256)
k_cp_wmma(const float* __restrict__ x,
          const float* __restrict__ TS,
          const float* __restrict__ TC,
          const float* __restrict__ SC,
          float* __restrict__ out) {
    const int n     = blockIdx.x / 48;
    const int chunk = blockIdx.x % 48;
    const int j0    = chunk * 256;
    const int tid   = threadIdx.x;

    // Fragment-ready LDS: [tile][frag][lane][16 halves contiguous]
    __shared__ __align__(32) _Float16 Asm[8  * 2 * 32 * 16];  // 16 KB
    __shared__ __align__(32) _Float16 Bsm[16 * 2 * 32 * 16];  // 32 KB

    // ---- stage A fragments: A[c][K] = SC[n,c,K], 16-bit A 16x32 ISA layout
    for (int e = tid; e < 8192; e += 256) {
        const int j  = e & 15;
        const int l  = (e >> 4) & 31;
        const int f  = (e >> 9) & 1;
        const int ct = e >> 10;
        const int half_ = l >> 4;
        const int v  = j >> 1, lo = j & 1;
        const int k32 = (v < 4) ? (half_ * 8 + (v << 1) + lo)
                                : (16 + half_ * 8 + ((v - 4) << 1) + lo);
        const int K = f * 32 + k32;
        const int c = (ct << 4) + (l & 15);
        Asm[e] = (_Float16)SC[((n << 7) + c) * RR + K];
    }
    // ---- stage B fragments: B'[K][jl] = TS[n,K,(j0+jl)/12] * TC[n,(j0+jl)%12,K]
    for (int e = tid; e < 16384; e += 256) {
        const int j  = e & 15;
        const int l  = (e >> 4) & 31;
        const int f  = (e >> 9) & 1;
        const int st = e >> 10;
        const int K  = f * 32 + ((j < 8) ? (((l >> 4) << 3) + j) : (16 + ((l >> 4) << 3) + (j - 8)));
        const int jl = (st << 4) + (l & 15);
        const int jg = j0 + jl;
        const int s  = jg / 12;
        const int t  = jg - s * 12;
        const float bv = TS[((n << 6) + K) * HWp + s] * TC[((n * TT + t) << 6) + K];
        Bsm[e] = (_Float16)bv;
    }
    __syncthreads();

    const int wave = tid >> 5;       // c-tile index 0..7
    const int lane = tid & 31;
    const v16h* Af = (const v16h*)Asm;
    const v16h* Bf = (const v16h*)Bsm;

    const v16h a0 = Af[(wave * 2 + 0) * 32 + lane];
    const v16h a1 = Af[(wave * 2 + 1) * 32 + lane];

    const int mbase = (wave << 4) + ((lane >> 4) << 3); // c = mbase + d
    const int ncol  = lane & 15;

    for (int st = 0; st < 16; ++st) {
        const v16h b0 = Bf[(st * 2 + 0) * 32 + lane];
        const v16h b1 = Bf[(st * 2 + 1) * 32 + lane];
        v8f acc = {};
        acc = __builtin_amdgcn_wmma_f32_16x16x32_f16(false, a0, false, b0,
                                                     (short)0, acc, false, false);
        acc = __builtin_amdgcn_wmma_f32_16x16x32_f16(false, a1, false, b1,
                                                     (short)0, acc, false, false);
        const int jg = j0 + (st << 4) + ncol;
#pragma unroll
        for (int d = 0; d < 8; ++d) {
            const int c = mbase + d;
            const size_t idx = ((size_t)((n << 7) + c)) * JTOT + jg;
            const float v = acc[d] * x[idx];     // coalesced: lanes step jg by 1
            out[idx] = v > 0.f ? v : 0.f;
        }
    }
}

// ---------------------------------------------------------------------------
extern "C" void kernel_launch(void* const* d_in, const int* in_sizes, int n_in,
                              void* d_out, int out_size, void* d_ws, size_t ws_size,
                              hipStream_t stream) {
    const float* x    = (const float*)d_in[0];
    const float* ts_w = (const float*)d_in[1];
    const float* ts_b = (const float*)d_in[2];
    const float* tc_w = (const float*)d_in[3];
    const float* tc_b = (const float*)d_in[4];
    const float* sc_w = (const float*)d_in[5];
    const float* sc_b = (const float*)d_in[6];
    float* out = (float*)d_out;

    // carve workspace (all fully written before read; no init needed)
    float* pooled = (float*)d_ws;                  // N*T*C      = 49152
    float* ts_in  = pooled + NN * TT * CC;         // N*T*HW     = 393216
    float* TS     = ts_in  + NN * TT * HWp;        // N*R*HW     = 2097152
    float* TC     = TS     + NN * RR * HWp;        // N*T*R      = 24576
    float* SC     = TC     + NN * TT * RR;         // N*C*R      = 262144

    k_pooled  <<<NN * CC,        384, 0, stream>>>(x, pooled);
    k_tsin    <<<NN * 32,        384, 0, stream>>>(x, ts_in);
    k_conv_sig<<<NN * RR,        256, 0, stream>>>(ts_in, ts_w, ts_b, TS);
    k_tc      <<<(NN*TT*RR)/256, 256, 0, stream>>>(pooled, tc_w, tc_b, TC);
    k_sc      <<<(NN*CC*RR)/256, 256, 0, stream>>>(pooled, sc_w, sc_b, SC);
    k_cp_wmma <<<NN * 48,        256, 0, stream>>>(x, TS, TC, SC, out);
}
